// TemporalGraphAttention_75557064671731
// MI455X (gfx1250) — matrix operand order repaired and checked by reading, not compiled, not run on hardware
//
#include <hip/hip_runtime.h>
#include <hip/hip_bf16.h>
#include <stdint.h>

#define N_NODES 32768
#define NGRAPH  64
#define NPG     512
#define D_IN    128
#define HID     64
#define HEADS   4
#define D_OUT   128
#define DH      256
#define NEG_SLOPE 0.2f
#define LN_EPS  1e-5f

typedef __attribute__((ext_vector_type(16))) _Float16 v16h;
typedef __attribute__((ext_vector_type(8)))  float    v8f;
typedef __attribute__((ext_vector_type(4)))  int      v4i;

union V16H { uint4 u[2]; v16h v; };
union V8F  { v8f v; float f[8]; };

// ---------------------------------------------------------------------------
// Async global->LDS staging (gfx1250 GLOBAL_LOAD_ASYNC_TO_LDS_B128, ASYNCcnt)
// with synchronous fallback when the builtin is unavailable.
// ---------------------------------------------------------------------------
#if __has_builtin(__builtin_amdgcn_global_load_async_to_lds_b128) && \
    __has_builtin(__builtin_amdgcn_s_wait_asynccnt)
#define USE_ASYNC_LDS 1
#else
#define USE_ASYNC_LDS 0
#endif

__device__ __forceinline__ void stage16(void* lds_dst, const void* gsrc) {
#if USE_ASYNC_LDS
  __builtin_amdgcn_global_load_async_to_lds_b128(
      (__attribute__((address_space(1))) v4i*)(gsrc),
      (__attribute__((address_space(3))) v4i*)(lds_dst), 0, 0);
#else
  *reinterpret_cast<uint4*>(lds_dst) = *reinterpret_cast<const uint4*>(gsrc);
#endif
}

__device__ __forceinline__ void stage_wait() {
#if USE_ASYNC_LDS
  __builtin_amdgcn_s_wait_asynccnt(0);
#endif
}

// ---------------------------------------------------------------------------
// Fragment gathers matching the CDNA5 WMMA VGPR layouts (CDNA5 ISA)
// ---------------------------------------------------------------------------
__device__ __forceinline__ v16h frag16(const _Float16* p0, const _Float16* p1) {
  V16H t;
  t.u[0] = *reinterpret_cast<const uint4*>(p0);
  t.u[1] = *reinterpret_cast<const uint4*>(p1);
  return t.v;
}

// A fragment (16x32 f16), row-major source with `stride` halves per row.
// lane<16: M=lane, koff=0 ; lane>=16: M=lane-16, koff=8.
// halves 0-7 -> K=koff..koff+7 ; halves 8-15 -> K=16+koff..16+koff+7.
__device__ __forceinline__ v16h frag_a(const _Float16* A, int stride, int lane) {
  int m = lane & 15;
  int koff = (lane < 16) ? 0 : 8;
  const _Float16* base = A + (size_t)m * stride + koff;
  return frag16(base, base + 16);
}

// B fragment (32x16 f16) from an [n][k]-ordered buffer: element (k,n) at
// B[n*stride+k]; halves 0-7 -> K=koff..koff+7, halves 8-15 -> K=16+koff.. (two 16B loads).
__device__ __forceinline__ v16h frag_b_nk(const _Float16* B, int stride, int lane) {
  int n = lane & 15;
  int koff = (lane < 16) ? 0 : 8;
  const _Float16* base = B + (size_t)n * stride + koff;
  return frag16(base, base + 16);
}

// ---------------------------------------------------------------------------
// f16 WMMA GEMM with pre-swizzled weights.
// B is stored fragment-ready: Bsw[((k/32)*Nc + n)*32 + (k%32)], so a block's
// 64-column k-panel is one contiguous 4KB chunk and every B fragment is two
// ds_load_b128 ops (no strided gathers, no transpose in the hot loop).
// 256 threads = 8 waves; block tile 128(M) x 64(N); each wave: 16x64 strip.
// ---------------------------------------------------------------------------
__global__ __launch_bounds__(256)
void gemm_wmma(const _Float16* __restrict__ A, const _Float16* __restrict__ Bsw,
               const float* __restrict__ bias, _Float16* __restrict__ out16,
               float* __restrict__ out32, int M, int Nc, int K, int act) {
  __shared__ __align__(16) _Float16 As[128 * 32];   // [m][k]
  __shared__ __align__(16) _Float16 Bs[64 * 32];    // [n][k]
  int tid  = threadIdx.x;
  int lane = tid & 31;
  int wave = tid >> 5;
  int row0 = blockIdx.y * 128;
  int col0 = blockIdx.x * 64;
  V8F acc[4] = {};

  for (int k0 = 0; k0 < K; k0 += 32) {
    { // stage A tile 128x32: 256 threads x 2 b128 granules
      int r   = tid >> 1;
      int seg = (tid & 1) * 16;
      const _Float16* g = A + (size_t)(row0 + r) * K + k0 + seg;
      stage16(&As[r * 32 + seg], g);
      stage16(&As[r * 32 + seg + 8], g + 8);
    }
    { // stage B panel (contiguous 2048 halves): 1 b128 granule per thread
      const _Float16* bpanel = Bsw + ((size_t)(k0 >> 5) * Nc + col0) * 32;
      stage16(&Bs[tid * 8], bpanel + tid * 8);
    }
    stage_wait();
    __syncthreads();
    v16h af = frag_a(&As[wave * 16 * 32], 32, lane);
#pragma unroll
    for (int j = 0; j < 4; ++j) {
      v16h bf = frag_b_nk(Bs + (j * 16) * 32, 32, lane);
      acc[j].v = __builtin_amdgcn_wmma_f32_16x16x32_f16(false, af, false, bf,
                                                        (short)0, acc[j].v, false, false);
    }
    __syncthreads();
  }

  int nlo   = lane & 15;
  int rbase = row0 + wave * 16 + ((lane < 16) ? 0 : 8);
#pragma unroll
  for (int j = 0; j < 4; ++j) {
    int col  = col0 + j * 16 + nlo;
    float bv = bias ? bias[col] : 0.0f;
#pragma unroll
    for (int r = 0; r < 8; ++r) {
      float v = acc[j].f[r] + bv;
      if (act == 1) v = (v > 0.0f) ? v : (__expf(v) - 1.0f);   // ELU
      size_t idx = (size_t)(rbase + r) * Nc + col;
      if (out16) out16[idx] = (_Float16)v;
      if (out32) out32[idx] = v;
    }
  }
}

// ---------------------------------------------------------------------------
// Flash attention: one wave per (graph, head, 16-query tile). hd=64, T=512.
// ---------------------------------------------------------------------------
__global__ __launch_bounds__(32)
void flash_attn(const _Float16* __restrict__ qkv, _Float16* __restrict__ out) {
  __shared__ __align__(16) _Float16 Kt[32 * 64];   // [key][dim] == [n][k] for S
  __shared__ __align__(16) _Float16 VtT[64 * 32];  // [dim][key] == [n][k] for O
  __shared__ __align__(16) _Float16 Pt[16 * 32];   // [m][key-in-tile]
  int lane = threadIdx.x;
  int id = blockIdx.x;
  int qt = id & 31;            // NPG/16 tiles
  int hh = (id >> 5) & 3;
  int bg = id >> 7;
  int row0 = bg * NPG + qt * 16;
  const int ld = 3 * DH;
  int kcol = DH + hh * 64;
  int vcol = 2 * DH + hh * 64;

  // Q fragments for dim chunks [0,32) and [32,64)
  v16h aq0, aq1;
  {
    int m  = lane & 15;
    int ko = (lane < 16) ? 0 : 8;
    const _Float16* base = qkv + (size_t)(row0 + m) * ld + hh * 64;
    aq0 = frag16(base + ko, base + ko + 16);
    aq1 = frag16(base + 32 + ko, base + 32 + ko + 16);
  }
  float m_i[8], l_i[8];
#pragma unroll
  for (int r = 0; r < 8; ++r) { m_i[r] = -1e30f; l_i[r] = 0.0f; }
  V8F O[4] = {};

  for (int kt = 0; kt < NPG; kt += 32) {
    for (int i = lane; i < 32 * 8; i += 32) {        // K tile: async b128 granules
      int key = i >> 3;
      int c8  = (i & 7) * 8;
      stage16(&Kt[key * 64 + c8],
              qkv + (size_t)(bg * NPG + kt + key) * ld + kcol + c8);
    }
    for (int i = lane; i < 32 * 16; i += 32) {       // V tile, transposed store
      int key = i >> 4;
      int c4  = (i & 15) * 4;
      const _Float16* g = qkv + (size_t)(bg * NPG + kt + key) * ld + vcol + c4;
      _Float16 v0 = g[0], v1 = g[1], v2 = g[2], v3 = g[3];
      VtT[(c4 + 0) * 32 + key] = v0;
      VtT[(c4 + 1) * 32 + key] = v1;
      VtT[(c4 + 2) * 32 + key] = v2;
      VtT[(c4 + 3) * 32 + key] = v3;
    }
    stage_wait();
    __syncthreads();

    // S = Q @ K^T for two 16-key blocks
    V8F s0 = {}, s1 = {};
    v16h bk;
    bk = frag_b_nk(Kt + 0 * 64 + 0,  64, lane);
    s0.v = __builtin_amdgcn_wmma_f32_16x16x32_f16(false, aq0, false, bk, (short)0, s0.v, false, false);
    bk = frag_b_nk(Kt + 0 * 64 + 32, 64, lane);
    s0.v = __builtin_amdgcn_wmma_f32_16x16x32_f16(false, aq1, false, bk, (short)0, s0.v, false, false);
    bk = frag_b_nk(Kt + 16 * 64 + 0,  64, lane);
    s1.v = __builtin_amdgcn_wmma_f32_16x16x32_f16(false, aq0, false, bk, (short)0, s1.v, false, false);
    bk = frag_b_nk(Kt + 16 * 64 + 32, 64, lane);
    s1.v = __builtin_amdgcn_wmma_f32_16x16x32_f16(false, aq1, false, bk, (short)0, s1.v, false, false);

    int n = lane & 15;
#pragma unroll
    for (int r = 0; r < 8; ++r) {      // online softmax per row (r + 8*(lane>=16))
      float a0 = s0.f[r] * 0.125f;     // 1/sqrt(64)
      float a1 = s1.f[r] * 0.125f;
      float mx = fmaxf(a0, a1);
#pragma unroll
      for (int off = 8; off >= 1; off >>= 1) mx = fmaxf(mx, __shfl_xor(mx, off, 32));
      float mnew = fmaxf(m_i[r], mx);
      float corr = __expf(m_i[r] - mnew);
      float p0 = __expf(a0 - mnew);
      float p1 = __expf(a1 - mnew);
      float rs = p0 + p1;
#pragma unroll
      for (int off = 8; off >= 1; off >>= 1) rs += __shfl_xor(rs, off, 32);
      l_i[r] = l_i[r] * corr + rs;
      m_i[r] = mnew;
#pragma unroll
      for (int j = 0; j < 4; ++j) O[j].f[r] *= corr;
      int mrow = r + ((lane < 16) ? 0 : 8);
      Pt[mrow * 32 + n]      = (_Float16)p0;
      Pt[mrow * 32 + 16 + n] = (_Float16)p1;
    }
    __syncthreads();

    // O += P @ V   (P: 16x32 A-frag ; V: 32x64 as 4 B-frags)
    v16h ap = frag_a(Pt, 32, lane);
#pragma unroll
    for (int j = 0; j < 4; ++j) {
      v16h bv = frag_b_nk(VtT + (j * 16) * 32, 32, lane);
      O[j].v = __builtin_amdgcn_wmma_f32_16x16x32_f16(false, ap, false, bv,
                                                      (short)0, O[j].v, false, false);
    }
    __syncthreads();
  }

  int n = lane & 15;
  int rbase = (lane < 16) ? 0 : 8;
#pragma unroll
  for (int r = 0; r < 8; ++r) {
    float inv = 1.0f / l_i[r];
    int row = row0 + rbase + r;
#pragma unroll
    for (int j = 0; j < 4; ++j)
      out[(size_t)row * DH + hh * 64 + j * 16 + n] = (_Float16)(O[j].f[r] * inv);
  }
}

// ---------------------------------------------------------------------------
// GAT helper kernels
// ---------------------------------------------------------------------------
__global__ void gat_scores(const _Float16* __restrict__ h, const float* __restrict__ aS,
                           const float* __restrict__ aD, float* __restrict__ es,
                           float* __restrict__ ed, int H, int C) {
  int node = blockIdx.x;
  int w = threadIdx.x >> 5;
  int lane = threadIdx.x & 31;
  const _Float16* hp = h + (size_t)node * H * C + w * C;
  const float* as = aS + w * C;
  const float* ad = aD + w * C;
  float ss = 0.f, sd = 0.f;
  for (int c = lane; c < C; c += 32) {
    float v = (float)hp[c];
    ss += v * as[c];
    sd += v * ad[c];
  }
  for (int off = 16; off >= 1; off >>= 1) {
    ss += __shfl_xor(ss, off, 32);
    sd += __shfl_xor(sd, off, 32);
  }
  if (lane == 0) { es[(size_t)node * H + w] = ss; ed[(size_t)node * H + w] = sd; }
}

__device__ __forceinline__ void atomicMaxF(float* addr, float val) {
  if (val >= 0.0f) atomicMax(reinterpret_cast<int*>(addr), __float_as_int(val));
  else             atomicMin(reinterpret_cast<unsigned int*>(addr), __float_as_uint(val));
}

__device__ __forceinline__ void edge_ends(const int* ei, int E, int e, int& s, int& d) {
  if (e < E) { s = ei[e]; d = ei[E + e]; } else { s = e - E; d = e - E; }  // self loops
}

__global__ void gat_edge_max(const int* __restrict__ ei, int E, int Etot,
                             const float* __restrict__ es, const float* __restrict__ ed,
                             float* __restrict__ mx, int H) {
  long long i = (long long)blockIdx.x * blockDim.x + threadIdx.x;
  if (i >= (long long)Etot * H) return;
  int e = (int)(i / H), hh = (int)(i % H);
  int s, d; edge_ends(ei, E, e, s, d);
  float v = es[(size_t)s * H + hh] + ed[(size_t)d * H + hh];
  v = (v >= 0.f) ? v : NEG_SLOPE * v;
  atomicMaxF(&mx[(size_t)d * H + hh], v);
}

__global__ void gat_edge_sum(const int* __restrict__ ei, int E, int Etot,
                             const float* __restrict__ es, const float* __restrict__ ed,
                             const float* __restrict__ mx, float* __restrict__ ssum,
                             float* __restrict__ pbuf, int H) {
  long long i = (long long)blockIdx.x * blockDim.x + threadIdx.x;
  if (i >= (long long)Etot * H) return;
  int e = (int)(i / H), hh = (int)(i % H);
  int s, d; edge_ends(ei, E, e, s, d);
  float v = es[(size_t)s * H + hh] + ed[(size_t)d * H + hh];
  v = (v >= 0.f) ? v : NEG_SLOPE * v;
  float p = __expf(v - mx[(size_t)d * H + hh]);
  pbuf[(size_t)e * H + hh] = p;
  atomicAdd(&ssum[(size_t)d * H + hh], p);
}

__global__ __launch_bounds__(128)
void gat_scatter(const _Float16* __restrict__ h, const int* __restrict__ ei, int E, int Etot,
                 const float* __restrict__ pbuf, const float* __restrict__ ssum,
                 float* __restrict__ agg, int H, int DHloc) {
  int e = blockIdx.x * 4 + (threadIdx.x >> 5);
  if (e >= Etot) return;
  int lane = threadIdx.x & 31;
  int s, d; edge_ends(ei, E, e, s, d);
  int Cp = DHloc / H;
  for (int ch = lane; ch < DHloc; ch += 32) {
    int hh = ch / Cp;
    float alpha = pbuf[(size_t)e * H + hh] / ssum[(size_t)d * H + hh];
    atomicAdd(&agg[(size_t)d * DHloc + ch], (float)h[(size_t)s * DHloc + ch] * alpha);
  }
}

__global__ void bias_act(const float* __restrict__ agg, const float* __restrict__ bias,
                         _Float16* __restrict__ out16, float* __restrict__ out32,
                         int DHloc, long long total, int act) {
  long long i = (long long)blockIdx.x * blockDim.x + threadIdx.x;
  if (i >= total) return;
  int c = (int)(i % DHloc);
  float v = agg[i] + bias[c];
  if (act == 1) v = (v > 0.f) ? v : (__expf(v) - 1.f);
  if (out16) out16[i] = (_Float16)v;
  if (out32) out32[i] = v;
}

// ---------------------------------------------------------------------------
// Misc elementwise kernels
// ---------------------------------------------------------------------------
__global__ void fill_f32(float* __restrict__ p, float v, long long n) {
  long long i = (long long)blockIdx.x * blockDim.x + threadIdx.x;
  if (i < n) p[i] = v;
}

__global__ void conv_f16(const float* __restrict__ in, _Float16* __restrict__ out, long long n) {
  long long i = (long long)blockIdx.x * blockDim.x + threadIdx.x;
  if (i < n) out[i] = (_Float16)in[i];
}

// Weight swizzle from row-major [K][Nc] f32:
// out[((k/32)*Nc + n)*32 + (k%32)] = in[k*Nc + n]
__global__ void conv_sw(const float* __restrict__ in, _Float16* __restrict__ out, int K, int Nc) {
  long long o = (long long)blockIdx.x * blockDim.x + threadIdx.x;
  if (o >= (long long)K * Nc) return;
  int kp = (int)(o / ((long long)Nc * 32));
  int r  = (int)(o % ((long long)Nc * 32));
  int n  = r >> 5;
  int kk = r & 31;
  out[o] = (_Float16)in[(size_t)(kp * 32 + kk) * Nc + n];
}

// Weight swizzle from [Nc][K] f32 (implicit transpose, for Wi / Wo):
// out[((k/32)*Nc + n)*32 + (k%32)] = in[n*K + k]
__global__ void conv_sw_t(const float* __restrict__ in, _Float16* __restrict__ out, int K, int Nc) {
  long long o = (long long)blockIdx.x * blockDim.x + threadIdx.x;
  if (o >= (long long)K * Nc) return;
  int kp = (int)(o / ((long long)Nc * 32));
  int r  = (int)(o % ((long long)Nc * 32));
  int n  = r >> 5;
  int kk = r & 31;
  out[o] = (_Float16)in[(size_t)n * K + kp * 32 + kk];
}

__global__ __launch_bounds__(128)
void add_ln(const _Float16* __restrict__ a, const _Float16* __restrict__ b,
            const float* __restrict__ gamma, const float* __restrict__ beta,
            _Float16* __restrict__ out) {
  int node = blockIdx.x * 4 + (threadIdx.x >> 5);
  int lane = threadIdx.x & 31;
  const _Float16* ap = a + (size_t)node * DH;
  const _Float16* bp = b + (size_t)node * DH;
  float vals[8], sum = 0.f, sq = 0.f;
#pragma unroll
  for (int i = 0; i < 8; ++i) {
    int c = lane + 32 * i;
    float v = (float)ap[c] + (float)bp[c];
    vals[i] = v; sum += v; sq += v * v;
  }
  for (int off = 16; off >= 1; off >>= 1) {
    sum += __shfl_xor(sum, off, 32);
    sq  += __shfl_xor(sq,  off, 32);
  }
  float mean = sum * (1.0f / DH);
  float var  = sq * (1.0f / DH) - mean * mean;
  float inv  = rsqrtf(var + LN_EPS);
  _Float16* op = out + (size_t)node * DH;
#pragma unroll
  for (int i = 0; i < 8; ++i) {
    int c = lane + 32 * i;
    op[c] = (_Float16)((vals[i] - mean) * inv * gamma[c] + beta[c]);
  }
}

// ---------------------------------------------------------------------------
// Host orchestration
// ---------------------------------------------------------------------------
extern "C" void kernel_launch(void* const* d_in, const int* in_sizes, int n_in,
                              void* d_out, int out_size, void* d_ws, size_t ws_size,
                              hipStream_t stream) {
  (void)n_in; (void)out_size; (void)ws_size;
  const float* x    = (const float*)d_in[0];
  const int*   ei   = (const int*)d_in[1];
  const float* W1   = (const float*)d_in[3];
  const float* aS1  = (const float*)d_in[4];
  const float* aD1  = (const float*)d_in[5];
  const float* b1   = (const float*)d_in[6];
  const float* W2   = (const float*)d_in[7];
  const float* aS2  = (const float*)d_in[8];
  const float* aD2  = (const float*)d_in[9];
  const float* b2   = (const float*)d_in[10];
  const float* W3   = (const float*)d_in[11];
  const float* aS3  = (const float*)d_in[12];
  const float* aD3  = (const float*)d_in[13];
  const float* b3   = (const float*)d_in[14];
  const float* Wi   = (const float*)d_in[15];
  const float* bi   = (const float*)d_in[16];
  const float* Wo   = (const float*)d_in[17];
  const float* bo   = (const float*)d_in[18];
  const float* gmm  = (const float*)d_in[19];
  const float* beta = (const float*)d_in[20];

  int E = in_sizes[1] / 2;
  int Etot = E + N_NODES;

  char* base = (char*)d_ws;
  size_t off = 0;
  auto alloc = [&](size_t bytes) -> void* {
    off = (off + 255) & ~(size_t)255;
    void* p = base + off;
    off += bytes;
    return p;
  };
  _Float16* x16    = (_Float16*)alloc((size_t)N_NODES * D_IN * 2);
  _Float16* w1h    = (_Float16*)alloc((size_t)D_IN * DH * 2);       // swizzled
  _Float16* w2h    = (_Float16*)alloc((size_t)DH * DH * 2);         // swizzled
  _Float16* w3h    = (_Float16*)alloc((size_t)DH * D_OUT * 2);      // swizzled
  _Float16* wiT    = (_Float16*)alloc((size_t)DH * 3 * DH * 2);     // swizzled (transposed)
  _Float16* woT    = (_Float16*)alloc((size_t)DH * DH * 2);         // swizzled (transposed)
  _Float16* h16    = (_Float16*)alloc((size_t)N_NODES * DH * 2);    // GAT pre-agg feats / x2a
  float*    agg    = (float*)   alloc((size_t)N_NODES * DH * 4);    // GAT aggregation
  _Float16* act16  = (_Float16*)alloc((size_t)N_NODES * DH * 2);    // x1 / attended
  _Float16* x2h    = (_Float16*)alloc((size_t)N_NODES * DH * 2);    // x2 (residual)
  _Float16* qkv16  = (_Float16*)alloc((size_t)N_NODES * 3 * DH * 2);
  _Float16* attn16 = (_Float16*)alloc((size_t)N_NODES * DH * 2);    // attn out / h3
  float*    es     = (float*)alloc((size_t)N_NODES * HEADS * 4);
  float*    ed     = (float*)alloc((size_t)N_NODES * HEADS * 4);
  float*    mx     = (float*)alloc((size_t)N_NODES * HEADS * 4);
  float*    ssum   = (float*)alloc((size_t)N_NODES * HEADS * 4);
  float*    pbuf   = (float*)alloc((size_t)Etot * HEADS * 4);

  auto cdiv = [](long long a, long long b) { return (int)((a + b - 1) / b); };

  // precision conversion: activations f32->f16; weights f32 -> fragment-ready
  // swizzled f16 (Wi/Wo with implicit transpose)
  conv_f16<<<cdiv((long long)N_NODES * D_IN, 256), 256, 0, stream>>>(x, x16, (long long)N_NODES * D_IN);
  conv_sw  <<<cdiv(D_IN * DH, 256), 256, 0, stream>>>(W1, w1h, D_IN, DH);
  conv_sw  <<<cdiv(DH * DH, 256), 256, 0, stream>>>(W2, w2h, DH, DH);
  conv_sw  <<<cdiv(DH * D_OUT, 256), 256, 0, stream>>>(W3, w3h, DH, D_OUT);
  conv_sw_t<<<cdiv(3 * DH * DH, 256), 256, 0, stream>>>(Wi, wiT, DH, 3 * DH);
  conv_sw_t<<<cdiv(DH * DH, 256), 256, 0, stream>>>(Wo, woT, DH, DH);

  auto gemm = [&](const _Float16* A, const _Float16* Bm, const float* bias,
                  _Float16* o16, float* o32, int M, int Nc, int K, int act) {
    dim3 g(Nc / 64, M / 128);
    gemm_wmma<<<g, 256, 0, stream>>>(A, Bm, bias, o16, o32, M, Nc, K, act);
  };

  auto gat = [&](const _Float16* h, const float* aS, const float* aD, const float* bias,
                 int H, int C, float* aggbuf, _Float16* o16, float* o32, int act) {
    int DHloc = H * C;
    gat_scores<<<N_NODES, 32 * H, 0, stream>>>(h, aS, aD, es, ed, H, C);
    fill_f32<<<cdiv((long long)N_NODES * H, 256), 256, 0, stream>>>(mx, -1e30f, (long long)N_NODES * H);
    fill_f32<<<cdiv((long long)N_NODES * H, 256), 256, 0, stream>>>(ssum, 0.f, (long long)N_NODES * H);
    gat_edge_max<<<cdiv((long long)Etot * H, 256), 256, 0, stream>>>(ei, E, Etot, es, ed, mx, H);
    gat_edge_sum<<<cdiv((long long)Etot * H, 256), 256, 0, stream>>>(ei, E, Etot, es, ed, mx, ssum, pbuf, H);
    fill_f32<<<cdiv((long long)N_NODES * DHloc, 256), 256, 0, stream>>>(aggbuf, 0.f, (long long)N_NODES * DHloc);
    gat_scatter<<<cdiv(Etot, 4), 128, 0, stream>>>(h, ei, E, Etot, pbuf, ssum, aggbuf, H, DHloc);
    bias_act<<<cdiv((long long)N_NODES * DHloc, 256), 256, 0, stream>>>(
        aggbuf, bias, o16, o32, DHloc, (long long)N_NODES * DHloc, act);
  };

  // GAT layer 1: x @ W1 -> attention aggregate -> +b1 -> ELU
  gemm(x16, w1h, nullptr, h16, nullptr, N_NODES, DH, D_IN, 0);
  gat(h16, aS1, aD1, b1, HEADS, HID, agg, act16, nullptr, 1);

  // GAT layer 2
  gemm(act16, w2h, nullptr, h16, nullptr, N_NODES, DH, DH, 0);
  gat(h16, aS2, aD2, b2, HEADS, HID, agg, x2h, nullptr, 1);

  // MHA: qkv = x2 @ Wi^T + bi ; flash attention ; proj = o @ Wo^T + bo
  gemm(x2h, wiT, bi, qkv16, nullptr, N_NODES, 3 * DH, DH, 0);
  flash_attn<<<NGRAPH * HEADS * (NPG / 16), 32, 0, stream>>>(qkv16, attn16);
  gemm(attn16, woT, bo, act16, nullptr, N_NODES, DH, DH, 0);

  // x2a = LayerNorm(attended + x2)
  add_ln<<<N_NODES / 4, 128, 0, stream>>>(act16, x2h, gmm, beta, h16);

  // GAT layer 3: x2a @ W3 -> aggregate -> +b3 (f32 output to d_out)
  gemm(h16, w3h, nullptr, attn16, nullptr, N_NODES, D_OUT, DH, 0);
  float* outp = (float*)d_out;
  gat(attn16, aS3, aD3, b3, 1, D_OUT, outp, nullptr, outp, 0);
}
